// GenerativeModel_60868276519628
// MI455X (gfx1250) — matrix operand order repaired and checked by reading, not compiled, not run on hardware
//
#include <hip/hip_runtime.h>

#define NN 8192
#define BB 64
#define LL 48
#define EE 65536
#define HH 128
#define RR 8
#define VAA 64
#define KC 1152   // RGCN concat K = R*H + H
#define KN 256    // node-MLP concat K = 2*H

typedef __attribute__((ext_vector_type(16))) __bf16 v16bf;
typedef __attribute__((ext_vector_type(8)))  float  v8f;

__device__ __forceinline__ unsigned short f2bf(float f) {
  union { float f; unsigned int u; } cv; cv.f = f;
  unsigned int u = cv.u;
  u += 0x7FFFu + ((u >> 16) & 1u);        // round-to-nearest-even bf16
  return (unsigned short)(u >> 16);
}

// ---------------- weight packing (bf16, B-fragment lane order) ----------------
// packed index = (((kt*8 + wave)*32 + lane)*16 + e);  k = kt*32 + 8*(lane>>4) + (e&7) + 2*(e&8)
// col = wave*16 + (lane&15)
__global__ void pack_rgcn_w(const float* __restrict__ rel, const float* __restrict__ root,
                            unsigned short* __restrict__ Wp) {
  int idx = blockIdx.x * blockDim.x + threadIdx.x;
  if (idx >= 3 * KC * HH) return;
  int li = idx / (KC * HH);
  int r = idx - li * (KC * HH);
  int e = r & 15, lane = (r >> 4) & 31, wave = (r >> 9) & 7, kt = r >> 12;
  int k = (kt << 5) + ((lane >> 4) << 3) + (e & 7) + ((e & 8) << 1);
  int col = (wave << 4) + (lane & 15);
  float v = (k < RR * HH) ? rel[((size_t)li * (RR * HH) + k) * HH + col]
                          : root[((size_t)li * HH + (k - RR * HH)) * HH + col];
  Wp[idx] = f2bf(v);
}

__global__ void pack_linN_w(const float* __restrict__ linN_w, unsigned short* __restrict__ Wp) {
  int idx = blockIdx.x * blockDim.x + threadIdx.x;
  if (idx >= KN * HH) return;
  int e = idx & 15, lane = (idx >> 4) & 31, wave = (idx >> 9) & 7, kt = idx >> 12;
  int k = (kt << 5) + ((lane >> 4) << 3) + (e & 7) + ((e & 8) << 1);
  int col = (wave << 4) + (lane & 15);
  Wp[idx] = f2bf(linN_w[col * KN + k]);   // linN_w is [H][2H], out = concat @ W^T
}

// ---------------- node embedding gather ----------------
__global__ void embed_kernel(const int* __restrict__ nodeTypes,
                             const float* __restrict__ emb, float* __restrict__ x) {
  int idx = blockIdx.x * blockDim.x + threadIdx.x;   // N*H
  int n = idx >> 7, h = idx & 127;
  x[idx] = emb[nodeTypes[n] * HH + h];
}

// ---------------- per-(target,relation) edge counts ----------------
__global__ void count_kernel(const int* __restrict__ ei, const int* __restrict__ ea,
                             float* __restrict__ cnt) {
  int e = blockIdx.x * blockDim.x + threadIdx.x;
  if (e < EE) atomicAdd(&cnt[ei[EE + e] * RR + ea[e]], 1.0f);
}

// ---------------- edge message scatter: agg[dst,rel,:] += x[src,:] ----------------
__global__ void scatter_kernel(const int* __restrict__ ei, const int* __restrict__ ea,
                               const float* __restrict__ x, float* __restrict__ agg) {
  int idx = blockIdx.x * blockDim.x + threadIdx.x;
  int e = idx >> 7, h = idx & 127;
  if (e < EE) {
    int src = ei[e], dst = ei[EE + e], r = ea[e];
    atomicAdd(&agg[((size_t)dst * RR + r) * HH + h], x[src * HH + h]);
  }
}

// ---------------- RGCN layer GEMM: x_new = relu([agg/denom | x] @ [rel;root] + bias) ----------------
__global__ void __launch_bounds__(256)
rgcn_wmma_kernel(const float* __restrict__ x, const float* __restrict__ agg,
                 const float* __restrict__ cnt, const unsigned short* __restrict__ Wp,
                 const float* __restrict__ bias, float* __restrict__ xout) {
  __shared__ __align__(32) unsigned short As[(KC / 32) * 32 * 16];  // 16x1152 tile, A-frag order
  const int node0 = blockIdx.x * 16;
  const int tid = threadIdx.x;
  for (int idx = tid; idx < 16 * KC; idx += 256) {
    int m = idx / KC, k = idx - m * KC;
    int n = node0 + m;
    float v;
    if (k < RR * HH) {
      float d = fmaxf(cnt[n * RR + (k >> 7)], 1.0f);          // mean aggregation
      v = agg[(size_t)n * (RR * HH) + k] / d;
    } else {
      v = x[n * HH + (k - RR * HH)];
    }
    int kt = k >> 5, kk = k & 31;
    int half = (kk >> 3) & 1;
    int e = (kk & 7) | ((kk & 16) >> 1);                      // invert k = 8*half + e + (e&8)
    As[(((kt << 5) + (m + (half << 4))) << 4) + e] = f2bf(v);
  }
  __syncthreads();
  const int wave = tid >> 5, lane = tid & 31;
  v8f c = {};
  for (int kt = 0; kt < KC / 32; ++kt) {
    v16bf a = *(const v16bf*)(const void*)&As[((kt << 5) + lane) << 4];
    v16bf b = *(const v16bf*)(const void*)&Wp[((((kt << 3) + wave) << 5) + lane) << 4];
    c = __builtin_amdgcn_wmma_f32_16x16x32_bf16(false, a, false, b, (short)0, c, false, false);
  }
  const int l16 = lane & 15, half = lane >> 4;
  const int col = (wave << 4) | l16;
  const float bb = bias[col];
#pragma unroll
  for (int i = 0; i < 8; ++i) {
    int node = node0 + i + (half << 3);                       // C layout: M = i + 8*half
    xout[node * HH + col] = fmaxf(c[i] + bb, 0.0f);
  }
}

// ---------------- global mean pool ----------------
__global__ void pool_kernel(const float* __restrict__ x, const int* __restrict__ nodes_bs,
                            float* __restrict__ hG) {
  const int b = blockIdx.x, tid = threadIdx.x;                // 128 threads = H channels
  int nb = nodes_bs[b];
  int base = b * (NN / BB);                                   // bs is sorted, equal-size graphs
  float acc = 0.0f;
  for (int i = 0; i < nb; ++i) acc += x[(base + i) * HH + tid];
  hG[b * HH + tid] = acc / (float)nb;
}

// ---------------- action head: relu(hG @ linA^T + b) @ linAf^T + b ----------------
__global__ void action_kernel(const float* __restrict__ hG,
                              const float* __restrict__ linA_w, const float* __restrict__ linA_b,
                              const float* __restrict__ linAf_w, const float* __restrict__ linAf_b,
                              float* __restrict__ action_out) {
  __shared__ float hid[HH];
  const int b = blockIdx.x, tid = threadIdx.x;
  const float* g = hG + b * HH;
  float acc = linA_b[tid];
  const float* wr = linA_w + tid * HH;
  for (int h = 0; h < HH; ++h) acc += g[h] * wr[h];
  hid[tid] = fmaxf(acc, 0.0f);
  __syncthreads();
  if (tid < VAA) {
    float a = linAf_b[tid];
    const float* wf = linAf_w + tid * HH;
    for (int o = 0; o < HH; ++o) a += hid[o] * wf[o];
    action_out[b * VAA + tid] = a;
  }
}

// ---------------- sequence embeddings: out[b,t,:] = sum_i seq[i,t]*x[i,:] ----------------
__global__ void seq_embed_kernel(const float* __restrict__ seq, const float* __restrict__ x,
                                 float* __restrict__ out) {
  const int bt = blockIdx.x;
  const int b = bt / LL, tid = threadIdx.x;
  const int t = bt - b * LL;
  const int base = b * (NN / BB);
  float acc = 0.0f;
  for (int i = 0; i < NN / BB; ++i) {
    float s = seq[(base + i) * LL + t];
    acc += s * x[(base + i) * HH + tid];
  }
  out[bt * HH + tid] = acc;
}

// ---------------- GRU input-gate precompute: xW = input_gru @ w_ih^T + b_ih ----------------
__global__ void gru_xw_kernel(const float* __restrict__ seq_emb,
                              const float* __restrict__ emb_actions,
                              const int* __restrict__ action_input,
                              const float* __restrict__ w_ih, const float* __restrict__ b_ih,
                              float* __restrict__ xW) {
  __shared__ float in[HH];
  const int bt = blockIdx.x;
  const int b = bt / LL, tid = threadIdx.x;                   // 384 threads
  const int t = bt - b * LL;
  if (tid < HH)
    in[tid] = (t == 0) ? emb_actions[action_input[b] * HH + tid]
                       : seq_emb[(bt - 1) * HH + tid];
  __syncthreads();
  float acc = b_ih[tid];
  const float* wr = w_ih + tid * HH;
  for (int h = 0; h < HH; ++h) acc += in[h] * wr[h];
  xW[(size_t)bt * 3 * HH + tid] = acc;
}

// ---------------- serial GRU scan, one block per graph ----------------
__global__ void gru_scan_kernel(const float* __restrict__ hG, const float* __restrict__ xW,
                                const float* __restrict__ w_hh, const float* __restrict__ b_hh,
                                const int* __restrict__ len_seq, float* __restrict__ seq_out) {
  __shared__ float hprev[HH];
  __shared__ float hw[3 * HH];
  const int b = blockIdx.x, tid = threadIdx.x;                // 128 threads
  hprev[tid] = hG[b * HH + tid];
  const int len = len_seq[b];
  __syncthreads();
  for (int t = 0; t < LL; ++t) {
    for (int g = tid; g < 3 * HH; g += HH) {
      float acc = b_hh[g];
      const float* wr = w_hh + g * HH;
      for (int h = 0; h < HH; ++h) acc += hprev[h] * wr[h];
      hw[g] = acc;
    }
    __syncthreads();
    const float* xwp = xW + (size_t)(b * LL + t) * 3 * HH;
    float rg = 1.0f / (1.0f + __expf(-(xwp[tid] + hw[tid])));
    float zg = 1.0f / (1.0f + __expf(-(xwp[HH + tid] + hw[HH + tid])));
    float ng = tanhf(xwp[2 * HH + tid] + rg * hw[2 * HH + tid]);
    float hn = (1.0f - zg) * ng + zg * hprev[tid];
    __syncthreads();
    hprev[tid] = hn;
    seq_out[(size_t)(b * LL + t) * HH + tid] = (t < len) ? hn : 0.0f;  // pad_packed mask
    __syncthreads();
  }
}

// ---------------- node MLP: logits = (relu([seq_out[bs],x] @ linN^T + b)) @ linNf^T + b ----------------
__global__ void __launch_bounds__(256)
linN_wmma_kernel(const float* __restrict__ x, const float* __restrict__ seq_out,
                 const unsigned short* __restrict__ Wp, const float* __restrict__ linN_b,
                 const float* __restrict__ linNf_w, const float* __restrict__ linNf_b,
                 float* __restrict__ logits) {
  __shared__ __align__(32) unsigned short As[16 * KN];
  __shared__ float lacc[16];
  const int row0 = blockIdx.x * 16;                           // rows over N*L
  const int tid = threadIdx.x;
  if (tid < 16) lacc[tid] = 0.0f;
  for (int idx = tid; idx < 16 * KN; idx += 256) {
    int m = idx >> 8, k = idx & 255;
    int row = row0 + m;
    int n = row / LL, t = row - n * LL;
    float v;
    if (k < HH) {
      int b = n >> 7;                                         // bs[n] = n / 128
      v = seq_out[(size_t)(b * LL + t) * HH + k];
    } else {
      v = x[n * HH + (k - HH)];
    }
    int kt = k >> 5, kk = k & 31;
    int half = (kk >> 3) & 1;
    int e = (kk & 7) | ((kk & 16) >> 1);
    As[(((kt << 5) + (m + (half << 4))) << 4) + e] = f2bf(v);
  }
  __syncthreads();
  const int wave = tid >> 5, lane = tid & 31;
  v8f c = {};
#pragma unroll
  for (int kt = 0; kt < KN / 32; ++kt) {
    v16bf a = *(const v16bf*)(const void*)&As[((kt << 5) + lane) << 4];
    v16bf b = *(const v16bf*)(const void*)&Wp[((((kt << 3) + wave) << 5) + lane) << 4];
    c = __builtin_amdgcn_wmma_f32_16x16x32_bf16(false, a, false, b, (short)0, c, false, false);
  }
  const int l16 = lane & 15, half = lane >> 4;
  const int col = (wave << 4) | l16;
  const float bN = linN_b[col];
  const float wf = linNf_w[col];
#pragma unroll
  for (int i = 0; i < 8; ++i) {
    float v = fmaxf(c[i] + bN, 0.0f) * wf;                    // fused relu + final dot
    v += __shfl_xor(v, 1);
    v += __shfl_xor(v, 2);
    v += __shfl_xor(v, 4);
    v += __shfl_xor(v, 8);                                    // sum over 16 cols in half-wave
    if (l16 == 0) atomicAdd(&lacc[i + (half << 3)], v);       // 8 waves x 16 cols -> 128 cols
  }
  __syncthreads();
  if (tid < 16) logits[row0 + tid] = lacc[tid] + linNf_b[0];
}

// ---------------- per-graph softmax over nodes (per timestep) ----------------
__global__ void softmax_kernel(const float* __restrict__ logits, float* __restrict__ outp) {
  __shared__ float red[NN / BB];
  const int bt = blockIdx.x;
  const int b = bt / LL, tid = threadIdx.x;                   // 128 nodes per graph
  const int t = bt - b * LL;
  const int node = b * (NN / BB) + tid;
  float v = logits[node * LL + t];
  red[tid] = v; __syncthreads();
  for (int s = 64; s > 0; s >>= 1) {
    if (tid < s) red[tid] = fmaxf(red[tid], red[tid + s]);
    __syncthreads();
  }
  float m = red[0]; __syncthreads();
  float e = __expf(v - m);
  red[tid] = e; __syncthreads();
  for (int s = 64; s > 0; s >>= 1) {
    if (tid < s) red[tid] += red[tid + s];
    __syncthreads();
  }
  outp[node * LL + t] = e / red[0];
}

extern "C" void kernel_launch(void* const* d_in, const int* in_sizes, int n_in,
                              void* d_out, int out_size, void* d_ws, size_t ws_size,
                              hipStream_t stream) {
  (void)in_sizes; (void)n_in; (void)out_size; (void)ws_size;
  const int*   nodeTypes   = (const int*)d_in[0];
  const int*   edge_index  = (const int*)d_in[1];
  const int*   edge_attr   = (const int*)d_in[2];
  const int*   nodes_bs    = (const int*)d_in[4];
  const int*   len_seq     = (const int*)d_in[5];
  const float* seq_in      = (const float*)d_in[6];
  const int*   action_in   = (const int*)d_in[7];
  const float* emb_nodes   = (const float*)d_in[8];
  const float* emb_actions = (const float*)d_in[9];
  const float* rgcn_rel    = (const float*)d_in[10];
  const float* rgcn_root   = (const float*)d_in[11];
  const float* rgcn_bias   = (const float*)d_in[12];
  const float* gru_w_ih    = (const float*)d_in[13];
  const float* gru_w_hh    = (const float*)d_in[14];
  const float* gru_b_ih    = (const float*)d_in[15];
  const float* gru_b_hh    = (const float*)d_in[16];
  const float* linA_w      = (const float*)d_in[17];
  const float* linA_b      = (const float*)d_in[18];
  const float* linAf_w     = (const float*)d_in[19];
  const float* linAf_b     = (const float*)d_in[20];
  const float* linN_w      = (const float*)d_in[21];
  const float* linN_b      = (const float*)d_in[22];
  const float* linNf_w     = (const float*)d_in[23];
  const float* linNf_b     = (const float*)d_in[24];

  // workspace layout (all offsets 32B-aligned)
  float* xA      = (float*)d_ws;
  float* xB      = xA + (size_t)NN * HH;
  float* cnt     = xB + (size_t)NN * HH;
  float* agg     = cnt + (size_t)NN * RR;
  float* hG      = agg + (size_t)NN * RR * HH;
  float* seq_emb = hG + (size_t)BB * HH;
  float* xW      = seq_emb + (size_t)BB * LL * HH;
  float* seq_out = xW + (size_t)BB * LL * 3 * HH;
  float* logits  = seq_out + (size_t)BB * LL * HH;
  unsigned short* WpR = (unsigned short*)(logits + (size_t)NN * LL);
  unsigned short* WpN = WpR + (size_t)3 * KC * HH;

  hipMemsetAsync(cnt, 0, (size_t)NN * RR * sizeof(float), stream);
  pack_rgcn_w<<<(3 * KC * HH + 255) / 256, 256, 0, stream>>>(rgcn_rel, rgcn_root, WpR);
  pack_linN_w<<<(KN * HH + 255) / 256, 256, 0, stream>>>(linN_w, WpN);
  embed_kernel<<<(NN * HH) / 256, 256, 0, stream>>>(nodeTypes, emb_nodes, xA);
  count_kernel<<<EE / 256, 256, 0, stream>>>(edge_index, edge_attr, cnt);

  float* xin = xA; float* xout = xB;
  for (int li = 0; li < 3; ++li) {
    hipMemsetAsync(agg, 0, (size_t)NN * RR * HH * sizeof(float), stream);
    scatter_kernel<<<(EE * HH) / 256, 256, 0, stream>>>(edge_index, edge_attr, xin, agg);
    rgcn_wmma_kernel<<<NN / 16, 256, 0, stream>>>(xin, agg, cnt, WpR + (size_t)li * KC * HH,
                                                  rgcn_bias + li * HH, xout);
    float* tmp = xin; xin = xout; xout = tmp;
  }

  pool_kernel<<<BB, HH, 0, stream>>>(xin, nodes_bs, hG);
  action_kernel<<<BB, HH, 0, stream>>>(hG, linA_w, linA_b, linAf_w, linAf_b, (float*)d_out);
  seq_embed_kernel<<<BB * LL, HH, 0, stream>>>(seq_in, xin, seq_emb);
  gru_xw_kernel<<<BB * LL, 3 * HH, 0, stream>>>(seq_emb, emb_actions, action_in,
                                                gru_w_ih, gru_b_ih, xW);
  gru_scan_kernel<<<BB, HH, 0, stream>>>(hG, xW, gru_w_hh, gru_b_hh, len_seq, seq_out);
  linN_wmma_kernel<<<(NN * LL) / 16, 256, 0, stream>>>(xin, seq_out, WpN, linN_b,
                                                       linNf_w, linNf_b, logits);
  softmax_kernel<<<BB * LL, HH, 0, stream>>>(logits, (float*)d_out + BB * VAA);
}
